// EquivariantFusionBottleneck_90941637526148
// MI455X (gfx1250) — matrix operand order repaired and checked by reading, not compiled, not run on hardware
//
#include <hip/hip_runtime.h>
#include <cmath>

typedef __bf16 bf16;
typedef __bf16 bf16x16 __attribute__((ext_vector_type(16)));
typedef __bf16 bf16x8  __attribute__((ext_vector_type(8)));
typedef __bf16 bf16x4  __attribute__((ext_vector_type(4)));
typedef float  v8f     __attribute__((ext_vector_type(8)));

#define TILE 16
#define DIN  480
#define DSC  384

// planar sub-offsets (in bf16 elements) inside the 16*480 x-staging buffer
#define XS_OFF 0       // 16 x 128, row stride 128
#define XV_OFF 2048    // 3 planes of 16 x 64, row stride 64
#define XT_OFF 5120    // 5 planes of 16 x 32, row stride 32

// packed-fragment offsets (bf16 elements) inside d_ws. Each fragment = 512 bf16.
#define OFF_WL0   0         // 4k * (KC4*NT8)*512 = 65536
#define OFF_WL1   65536     // 4k * (KC2*NT4)*512 = 16384
#define OFF_WL2   81920     // 4k * (KC1*NT2)*512 = 4096
#define OFF_TP0   86016     // (KC2*NT4)*512 = 4096
#define OFF_TP1   90112     // 4096
#define OFF_WS    94208     // 4k * (KC6*NT16)*512 = 196608
#define OFF_WV1O  290816    // 4k * (KC2*NT4)*512 = 16384
#define OFF_WM1   307200    // (KC12*NT4)*512 = 24576   (total 331776 bf16 = 648 KB)

__device__ __forceinline__ v8f wmma_bf16(bf16x16 a, bf16x16 b, v8f c) {
  return __builtin_amdgcn_wmma_f32_16x16x32_bf16(false, a, false, b, (short)0, c, false, false);
}

__device__ __forceinline__ bf16x16 load_B(const bf16* __restrict__ p, int frag, int lane) {
  return *(const bf16x16*)(p + (size_t)frag * 512 + lane * 16);
}

// A-operand, contiguous K: two 16B LDS vector loads.
// Layout: lane holds row (lane&15); half h -> k = kc*32 + 8*(lane>>4) + h (+8 if h>=8).
__device__ __forceinline__ bf16x16 load_A_c(const bf16* __restrict__ buf, int RS, int kc, int lane) {
  const int mr = lane & 15, hi = lane >> 4;
  const bf16* p = buf + mr * RS + kc * 32 + 8 * hi;
  bf16x8 lo = *(const bf16x8*)p;
  bf16x8 hh = *(const bf16x8*)(p + 16);
  return __builtin_shufflevector(lo, hh, 0,1,2,3,4,5,6,7,8,9,10,11,12,13,14,15);
}

// Pack one f32 weight matrix (batch x Kdim x Ndim) into WMMA bf16 B-fragments, scale folded.
// B layout: lane (l) holds column n = nt*16 + (l&15); half h -> k = kc*32 + (h&7) + 8*(l>>4) + 16*(h>>3).
__global__ __launch_bounds__(512) void pack_b(const float* __restrict__ src, bf16* __restrict__ dst,
                                              int Kdim, int Ndim, int KC, int NT,
                                              long batchStride, float scale) {
  const int fpb  = KC * NT;
  const int frag = blockIdx.x;
  const int b = frag / fpb, rem = frag % fpb;
  const int kc = rem / NT, nt = rem % NT;
  const int t = threadIdx.x, l = t >> 4, h = t & 15;
  const int ks = kc * 32 + (h & 7) + 8 * (l >> 4) + 16 * (h >> 3);
  const int ns = nt * 16 + (l & 15);
  float v = 0.f;
  if (ks < Kdim && ns < Ndim) v = src[(long)b * batchStride + (long)ks * Ndim + ns] * scale;
  dst[(size_t)frag * 512 + l * 16 + h] = (bf16)v;
}

__global__ __launch_bounds__(128) void fused_forward(
    const float* __restrict__ x0, const float* __restrict__ x1,
    const float* __restrict__ x2, const float* __restrict__ x3,
    const float* __restrict__ bs,   const float* __restrict__ Wv1e,
    const float* __restrict__ gamma, const float* __restrict__ beta,
    const float* __restrict__ bm1,  const float* __restrict__ Wm2,
    const float* __restrict__ bm2,  const bf16* __restrict__ pk,
    float* __restrict__ out_s, float* __restrict__ out_v, int N)
{
  __shared__ __attribute__((aligned(16))) bf16 sXbuf[TILE * DIN];  // planar x tile; reused as H
  __shared__ __attribute__((aligned(16))) bf16 sScat[TILE * 192];  // [s_k | n_k | p0 | pad0]
  __shared__ __attribute__((aligned(16))) bf16 sVplan[3 * TILE * 64]; // v_k planar (A-operand form)
  __shared__ __attribute__((aligned(16))) bf16 sVIA[TILE * 256];   // v interleaved4: row,col,(m0..m2,pad)
  __shared__ __attribute__((aligned(16))) bf16 sVIB[TILE * 256];
  __shared__ __attribute__((aligned(16))) bf16 sS[4][TILE * 256];
  __shared__ __attribute__((aligned(16))) bf16 sV[4][TILE * 384];
  __shared__ float sP0[16];
  __shared__ float sP1[48];
  __shared__ float sE[64];
  __shared__ float sRed[128], sRed2[128], sMu[16], sRstd[16];

  const int tid  = threadIdx.x;
  const int l    = tid & 31;   // lane
  const int w    = tid >> 5;   // wave id (0..3)
  const int hi   = l >> 4;
  const int base = blockIdx.x * TILE;

  // zero pad columns 160..191 of the concat buffer once (col 160 rewritten per-k with p0)
  for (int i = tid; i < TILE * 32; i += 128) {
    const int row = i >> 5, c = 160 + (i & 31);
    sScat[row * 192 + c] = (bf16)0.f;
  }

  bf16* viC = sVIA;   // current v_k, interleaved4 (consumed at k+1)
  bf16* viP = sVIB;   // previous v_k, interleaved4
  const float* xs[4] = {x0, x1, x2, x3};

  for (int k = 0; k < 4; ++k) {
    const bf16* pWl0 = pk + OFF_WL0  + k * 16384;
    const bf16* pWl1 = pk + OFF_WL1  + k * 4096;
    const bf16* pWl2 = pk + OFF_WL2  + k * 1024;
    const bf16* pWsk = pk + OFF_WS   + k * 49152;
    const bf16* pV1O = pk + OFF_WV1O + k * 4096;

    // ---- stage x tile: f32 global (b128 loads) -> de-interleaved bf16 planes in LDS ----
    {
      const float* xk = xs[k];
      for (int i = tid; i < TILE * (DIN / 4); i += 128) {
        const int row = i / (DIN / 4), c4 = i % (DIN / 4);
        int n = base + row; if (n >= N) n = N - 1;
        const float4 v = ((const float4*)(xk + (size_t)n * DIN))[c4];
        const float f4[4] = {v.x, v.y, v.z, v.w};
        #pragma unroll
        for (int e = 0; e < 4; ++e) {
          const int cc = c4 * 4 + e;
          int dst;
          if (cc < 128) {
            dst = XS_OFF + row * 128 + cc;                       // scalar block
          } else if (cc < 320) {
            const int r2 = cc - 128;                             // v block: col = 3*i + m
            dst = XV_OFF + (r2 % 3) * (TILE * 64) + row * 64 + (r2 / 3);
          } else {
            const int r2 = cc - 320;                             // T block: col = 5*i + m
            dst = XT_OFF + (r2 % 5) * (TILE * 32) + row * 32 + (r2 / 5);
          }
          sXbuf[dst] = (bf16)f4[e];
        }
      }
      if (k < 3 && tid < TILE) {
        int n = base + tid; if (n >= N) n = N - 1;
        __builtin_prefetch(xs[k + 1] + (size_t)n * DIN, 0, 0);
      }
    }
    __syncthreads();

    // v_k GEMM helper: A frags of plane m loaded once, reused across nt tiles
    auto do_vk = [&](int m, int nt0, int nt1) {
      const bf16x16 a0 = load_A_c(sXbuf + XV_OFF + m * (TILE * 64), 64, 0, l);
      const bf16x16 a1 = load_A_c(sXbuf + XV_OFF + m * (TILE * 64), 64, 1, l);
      for (int nt = nt0; nt < nt1; ++nt) {
        v8f acc = {};
        acc = wmma_bf16(a0, load_B(pWl1, nt, l), acc);
        acc = wmma_bf16(a1, load_B(pWl1, 4 + nt, l), acc);
        const int col = nt * 16 + (l & 15);
        #pragma unroll
        for (int r = 0; r < 8; ++r) {
          const int row = r + 8 * hi;
          const bf16 bv = (bf16)acc[r];
          sVplan[m * (TILE * 64) + row * 64 + col] = bv;   // A-operand form (this k)
          viC[row * 256 + col * 4 + m] = bv;               // elementwise form (next k)
        }
      }
    };

    // ---- phase 1: s_k (waves 0-1), v_k + |T_k| (waves 2-3) ----
    if (w < 2) {
      bf16x16 aS[4];
      #pragma unroll
      for (int kc = 0; kc < 4; ++kc) aS[kc] = load_A_c(sXbuf + XS_OFF, 128, kc, l);
      for (int j = 0; j < 4; ++j) {
        const int nt = w * 4 + j;
        v8f acc = {};
        #pragma unroll
        for (int kc = 0; kc < 4; ++kc) acc = wmma_bf16(aS[kc], load_B(pWl0, kc * 8 + nt, l), acc);
        const int col = nt * 16 + (l & 15);
        #pragma unroll
        for (int r = 0; r < 8; ++r) sScat[(r + 8 * hi) * 192 + col] = (bf16)acc[r];
      }
    } else {
      const int wq = w - 2;
      if (wq == 0) { do_vk(0, 0, 4); do_vk(1, 0, 2); }
      else         { do_vk(1, 2, 4); do_vk(2, 0, 4); }
      { // T_k: accumulate sum of squares over the 5 components, then sqrt -> n_k cols 128..159
        const int nt = wq;
        float sq[8] = {0.f,0.f,0.f,0.f,0.f,0.f,0.f,0.f};
        for (int m = 0; m < 5; ++m) {
          v8f acc = {};
          acc = wmma_bf16(load_A_c(sXbuf + XT_OFF + m * (TILE * 32), 32, 0, l),
                          load_B(pWl2, nt, l), acc);
          #pragma unroll
          for (int r = 0; r < 8; ++r) sq[r] += acc[r] * acc[r];
        }
        const int col = nt * 16 + (l & 15);
        #pragma unroll
        for (int r = 0; r < 8; ++r) sScat[(r + 8 * hi) * 192 + 128 + col] = (bf16)sqrtf(sq[r]);
      }
    }
    if (tid < 16) sP0[tid] = 0.f;
    if (tid < 48) sP1[tid] = 0.f;
    __syncthreads();

    // ---- phase 2: p0 = <u0, v_prev>, p1 = sum_j cross(u1, v_prev); scales folded in pTP* ----
    if (k > 0) {
      const int nt  = w;
      const int col = nt * 16 + (l & 15);
      v8f u0[3], u1[3];
      #pragma unroll
      for (int m = 0; m < 3; ++m) {
        v8f a0 = {}, a1 = {};
        #pragma unroll
        for (int kc = 0; kc < 2; ++kc) {
          const bf16x16 av = load_A_c(sVplan + m * (TILE * 64), 64, kc, l);
          a0 = wmma_bf16(av, load_B(pk + OFF_TP0, kc * 4 + nt, l), a0);
          a1 = wmma_bf16(av, load_B(pk + OFF_TP1, kc * 4 + nt, l), a1);
        }
        u0[m] = a0; u1[m] = a1;
      }
      #pragma unroll
      for (int r = 0; r < 8; ++r) {
        const int row = r + 8 * hi;
        const bf16x4 vv = *(const bf16x4*)(viP + row * 256 + col * 4);
        const float v0 = (float)vv[0], v1 = (float)vv[1], v2 = (float)vv[2];
        atomicAdd(&sP0[row], u0[0][r] * v0 + u0[1][r] * v1 + u0[2][r] * v2);
        atomicAdd(&sP1[row * 3 + 0], u1[1][r] * v2 - u1[2][r] * v1);
        atomicAdd(&sP1[row * 3 + 1], u1[2][r] * v0 - u1[0][r] * v2);
        atomicAdd(&sP1[row * 3 + 2], u1[0][r] * v1 - u1[1][r] * v0);
      }
    }
    __syncthreads();
    if (tid < 16) sScat[tid * 192 + 160] = (bf16)sP0[tid];
    __syncthreads();

    // ---- phase 3: s_tilde = concat @ Ws[k] + bs[k];  v1o;  v1e = p1 (x) Wv1e[k] ----
    {
      bf16x16 aC[6];
      #pragma unroll
      for (int kc = 0; kc < 6; ++kc) aC[kc] = load_A_c(sScat, 192, kc, l);
      for (int j = 0; j < 4; ++j) {
        const int nt = w * 4 + j;
        v8f acc = {};
        #pragma unroll
        for (int kc = 0; kc < 6; ++kc) acc = wmma_bf16(aC[kc], load_B(pWsk, kc * 16 + nt, l), acc);
        const int col = nt * 16 + (l & 15);
        const float bias = bs[k * 256 + col];
        #pragma unroll
        for (int r = 0; r < 8; ++r) sS[k][(r + 8 * hi) * 256 + col] = (bf16)(acc[r] + bias);
      }
    }
    {
      auto do_v1o = [&](int m, int nt0, int nt1) {
        const bf16x16 a0 = load_A_c(sVplan + m * (TILE * 64), 64, 0, l);
        const bf16x16 a1 = load_A_c(sVplan + m * (TILE * 64), 64, 1, l);
        for (int nt = nt0; nt < nt1; ++nt) {
          v8f acc = {};
          acc = wmma_bf16(a0, load_B(pV1O, nt, l), acc);
          acc = wmma_bf16(a1, load_B(pV1O, 4 + nt, l), acc);
          const int col = nt * 16 + (l & 15);
          #pragma unroll
          for (int r = 0; r < 8; ++r) sV[k][(r + 8 * hi) * 384 + 3 * col + m] = (bf16)acc[r];
        }
      };
      if      (w == 0) { do_v1o(0, 0, 3); }
      else if (w == 1) { do_v1o(0, 3, 4); do_v1o(1, 0, 2); }
      else if (w == 2) { do_v1o(1, 2, 4); do_v1o(2, 0, 1); }
      else             { do_v1o(2, 1, 4); }
    }
    for (int i = tid; i < TILE * 192; i += 128) {
      const int row = i / 192, rem = i % 192, ii = rem / 3, m = rem % 3;
      sV[k][row * 384 + 192 + rem] = (bf16)(sP1[row * 3 + m] * Wv1e[k * 64 + ii]);
    }

    { bf16* t = viC; viC = viP; viP = t; }
    __syncthreads();
  }

  // ---- phase 4: LayerNorm + MLP scorer (WMMA) + softmax over k + alpha-weighted fusion ----
  if (tid < 64) sE[tid] = 0.f;
  bf16* sH = sXbuf;                    // reuse x staging buffer for normalized rows (bf16)
  const int row8 = tid >> 3, sub = tid & 7;   // 8 threads per sample row
  for (int k = 0; k < 4; ++k) {
    const bf16* Sk = sS[k];
    const bf16* Vk = sV[k];
    float ps = 0.f, pss = 0.f;
    for (int c = sub; c < DSC; c += 8) {      // pass 1: raw score vector -> sH + stats
      float v;
      if (c < 256) v = (float)Sk[row8 * 256 + c];
      else {
        const int j3 = (c - 256) * 3;
        const float a = (float)Vk[row8 * 384 + j3 + 0];
        const float b = (float)Vk[row8 * 384 + j3 + 1];
        const float d = (float)Vk[row8 * 384 + j3 + 2];
        v = sqrtf(a * a + b * b + d * d);
      }
      sH[row8 * DSC + c] = (bf16)v;
      ps += v; pss += v * v;
    }
    sRed[tid] = ps; sRed2[tid] = pss;
    __syncthreads();
    if (sub == 0) {
      float s = 0.f, ss = 0.f;
      for (int j = 0; j < 8; ++j) { s += sRed[row8 * 8 + j]; ss += sRed2[row8 * 8 + j]; }
      const float mu = s / (float)DSC;
      sMu[row8] = mu; sRstd[row8] = rsqrtf(ss / (float)DSC - mu * mu + 1e-5f);
    }
    __syncthreads();
    const float mu = sMu[row8], rs = sRstd[row8];
    for (int c = sub; c < DSC; c += 8) {      // pass 2: affine in place
      const float v = (float)sH[row8 * DSC + c];
      sH[row8 * DSC + c] = (bf16)(((v - mu) * rs) * gamma[c] + beta[c]);
    }
    __syncthreads();
    { // e[:,k] = silu(H @ Wm1 + bm1) @ Wm2  via WMMA; wave w owns hidden cols [16w,16w+16)
      v8f acc = {};
      #pragma unroll
      for (int kc = 0; kc < 12; ++kc)
        acc = wmma_bf16(load_A_c(sH, DSC, kc, l), load_B(pk + OFF_WM1, kc * 4 + w, l), acc);
      const int col = w * 16 + (l & 15);
      const float b1 = bm1[col], w2 = Wm2[col];
      #pragma unroll
      for (int r = 0; r < 8; ++r) {
        const float x = acc[r] + b1;
        atomicAdd(&sE[(r + 8 * hi) * 4 + k], (x / (1.f + expf(-x))) * w2);
      }
    }
    __syncthreads();
  }
  if (tid < 16) {
    const float b2 = bm2[0];
    float e[4];
    #pragma unroll
    for (int k = 0; k < 4; ++k) e[k] = sE[tid * 4 + k] + b2;
    const float mx = fmaxf(fmaxf(e[0], e[1]), fmaxf(e[2], e[3]));
    float s = 0.f;
    #pragma unroll
    for (int k = 0; k < 4; ++k) { e[k] = expf(e[k] - mx); s += e[k]; }
    #pragma unroll
    for (int k = 0; k < 4; ++k) sE[tid * 4 + k] = e[k] / s;
  }
  __syncthreads();
  for (int i = tid; i < TILE * 256; i += 128) {
    const int row = i >> 8, c = i & 255;
    if (base + row < N) {
      float acc = 0.f;
      #pragma unroll
      for (int k = 0; k < 4; ++k) acc += sE[row * 4 + k] * (float)sS[k][row * 256 + c];
      out_s[(size_t)(base + row) * 256 + c] = acc;
    }
  }
  for (int i = tid; i < TILE * 384; i += 128) {
    const int row = i / 384, c = i % 384;
    if (base + row < N) {
      float acc = 0.f;
      #pragma unroll
      for (int k = 0; k < 4; ++k) acc += sE[row * 4 + k] * (float)sV[k][row * 384 + c];
      out_v[(size_t)(base + row) * 384 + c] = acc;
    }
  }
}

extern "C" void kernel_launch(void* const* d_in, const int* in_sizes, int n_in,
                              void* d_out, int out_size, void* d_ws, size_t ws_size,
                              hipStream_t stream) {
  (void)n_in; (void)out_size; (void)ws_size;
  const float* x0   = (const float*)d_in[0];
  const float* x1   = (const float*)d_in[1];
  const float* x2   = (const float*)d_in[2];
  const float* x3   = (const float*)d_in[3];
  const float* Wl0  = (const float*)d_in[4];
  const float* Wl1  = (const float*)d_in[5];
  const float* Wl2  = (const float*)d_in[6];
  const float* Wtp0 = (const float*)d_in[7];
  const float* Wtp1 = (const float*)d_in[8];
  const float* Ws   = (const float*)d_in[9];
  const float* bs   = (const float*)d_in[10];
  const float* Wv1o = (const float*)d_in[11];
  const float* Wv1e = (const float*)d_in[12];
  const float* gamma= (const float*)d_in[13];
  const float* beta = (const float*)d_in[14];
  const float* Wm1  = (const float*)d_in[15];
  const float* bm1  = (const float*)d_in[16];
  const float* Wm2  = (const float*)d_in[17];
  const float* bm2  = (const float*)d_in[18];

  bf16* pk = (bf16*)d_ws;                     // needs ~648 KB of workspace
  const int N = in_sizes[0] / DIN;

  const float inv0 = 0.08838834764831845f;    // 1/sqrt(128)
  const float inv1 = 0.125f;                  // 1/sqrt(64)
  const float inv2 = 0.17677669529663687f;    // 1/sqrt(32)
  const float p0s  = 0.009021097761f;         // 1/(sqrt(3)*64)
  const float p1s  = 0.011048543456f;         // 1/(sqrt(2)*64)

  pack_b<<<128, 512, 0, stream>>>(Wl0,  pk + OFF_WL0,  128, 128,  4,  8, 16384, inv0);
  pack_b<<< 32, 512, 0, stream>>>(Wl1,  pk + OFF_WL1,   64,  64,  2,  4,  4096, inv1);
  pack_b<<<  8, 512, 0, stream>>>(Wl2,  pk + OFF_WL2,   32,  32,  1,  2,  1024, inv2);
  pack_b<<<  8, 512, 0, stream>>>(Wtp0, pk + OFF_TP0,   64,  64,  2,  4,  4096, p0s);
  pack_b<<<  8, 512, 0, stream>>>(Wtp1, pk + OFF_TP1,   64,  64,  2,  4,  4096, p1s);
  pack_b<<<384, 512, 0, stream>>>(Ws,   pk + OFF_WS,   161, 256,  6, 16, 41216, 1.0f);
  pack_b<<< 32, 512, 0, stream>>>(Wv1o, pk + OFF_WV1O,  64,  64,  2,  4,  4096, inv1);
  pack_b<<< 48, 512, 0, stream>>>(Wm1,  pk + OFF_WM1,  384,  64, 12,  4, 24576, 1.0f);

  float* out_s = (float*)d_out;
  float* out_v = out_s + (size_t)N * 256;
  fused_forward<<<(N + TILE - 1) / TILE, 128, 0, stream>>>(
      x0, x1, x2, x3, bs, Wv1e, gamma, beta, bm1, Wm2, bm2, pk, out_s, out_v, N);
}
